// SampledSoftmax_20246475833762
// MI455X (gfx1250) — matrix operand order repaired and checked by reading, not compiled, not run on hardware
//
#include <hip/hip_runtime.h>
#include <hip/hip_bf16.h>

#define B_ 32
#define S_ 512
#define D_ 1024
#define V_ 50000
#define NS_ 1000
#define NPAD 1024               // padded per-batch metadata / gathered-row stride
#define STILE 64
#define XSTRIDE 1040            // bf16 elements, padded to de-correlate LDS banks
#define NEGBIG (-3.0e38f)

typedef __attribute__((ext_vector_type(16))) __bf16 v16bf;
typedef __attribute__((ext_vector_type(8)))  float  v8f;

union Frag16 { v16bf v; uint4 q[2]; };

// ---------------------------------------------------------------------------
// Kernel 1: padded metadata.  For n < NS:  sidP = sampled id,
//   beffP = softmax_b[sid] - log(q(sid)).  For n >= NS: sidP=-1, beffP=-3e38
//   so tail columns self-mask in the online logsumexp (no bounds checks).
// ---------------------------------------------------------------------------
__global__ void ssm_meta_kernel(const int* __restrict__ sampled_ids,
                                const float* __restrict__ softmax_b,
                                int* __restrict__ sidP,
                                float* __restrict__ beffP) {
    int j = blockIdx.x * blockDim.x + threadIdx.x;   // 0 .. B*NPAD-1
    int b = j >> 10;
    int n = j & (NPAD - 1);
    if (b < B_) {
        int   sid = -1;
        float be  = NEGBIG;
        if (n < NS_) {
            sid = sampled_ids[b * NS_ + n];
            float sf = (float)sid;
            float p  = __logf((sf + 2.0f) / (sf + 1.0f)) *
                       (1.0f / __logf((float)(V_ + 1)));
            be = softmax_b[sid] - __logf(p * (float)NS_);
        }
        sidP[j]  = sid;
        beffP[j] = be;
    }
}

// ---------------------------------------------------------------------------
// Kernel 2: one-time gather + fp32->bf16 convert of the sampled W rows.
//   Wg[b][n][k] (n padded to NPAD, tail rows zero).  This removes all
//   conversion VALU and halves W read traffic in the WMMA hot loop.
// Grid = B*NPAD blocks of 128 threads; 8 elements per thread.
// ---------------------------------------------------------------------------
__global__ void ssm_wconv_kernel(const int* __restrict__ sampled_ids,
                                 const float* __restrict__ W,
                                 __bf16* __restrict__ Wg) {
    int row = blockIdx.x;            // 0 .. B*NPAD-1
    int b   = row >> 10;
    int n   = row & (NPAD - 1);
    int t   = threadIdx.x;           // 0..127
    __bf16* dst = Wg + (size_t)row * D_ + t * 8;
    if (n < NS_) {
        const float* src = W + (size_t)sampled_ids[b * NS_ + n] * D_ + t * 8;
        float4 f0 = ((const float4*)src)[0];
        float4 f1 = ((const float4*)src)[1];
        dst[0] = (__bf16)f0.x; dst[1] = (__bf16)f0.y;
        dst[2] = (__bf16)f0.z; dst[3] = (__bf16)f0.w;
        dst[4] = (__bf16)f1.x; dst[5] = (__bf16)f1.y;
        dst[6] = (__bf16)f1.z; dst[7] = (__bf16)f1.w;
    } else {
#pragma unroll
        for (int i = 0; i < 8; ++i) dst[i] = (__bf16)0.0f;
    }
}

// ---------------------------------------------------------------------------
// Kernel 3: fused sampled-softmax. Grid = B * (S/STILE) blocks of 128 threads.
// Block: batch b, 64 s-rows (X tile bf16 in LDS). Each wave owns 16 s-columns;
// n-tiles of 16 pre-gathered bf16 W rows stream through
// v_wmma_f32_16x16x32_bf16 with an online logsumexp per s.
// Steady-state loop: 2x global_load_b128 + 2x ds_load_b128 + 1x wmma per K=32.
// ---------------------------------------------------------------------------
__launch_bounds__(128, 1)
__global__ void ssm_main_kernel(const float*  __restrict__ X,        // [B,S,D]
                                const int*    __restrict__ trueIds,  // [B,S]
                                const float*  __restrict__ W,        // [V,D]
                                const float*  __restrict__ bvec,     // [V]
                                const __bf16* __restrict__ Wg,       // [B,NPAD,D]
                                const int*    __restrict__ sidP,     // [B,NPAD]
                                const float*  __restrict__ beffP,    // [B,NPAD]
                                float* __restrict__ partial)         // [B*S/STILE]
{
    __shared__ __bf16 Xl[STILE * XSTRIDE];
    __shared__ float  tl[STILE];
    __shared__ float  wsum[4];

    const int bid   = blockIdx.x;
    const int b     = bid >> 3;            // S_/STILE == 8 tiles per batch
    const int stile = bid & 7;
    const int s0    = stile * STILE;
    const int t     = threadIdx.x;
    const int wave  = t >> 5;
    const int lane  = t & 31;
    const int half  = lane >> 4;
    const int l15   = lane & 15;

    // ---- stage X tile (64 x 1024 fp32 -> bf16 LDS), 512 elems/thread ----
    {
        int row = t >> 1;
        int c0  = (t & 1) * 512;
        const float* src = X + ((size_t)b * S_ + (size_t)(s0 + row)) * D_ + c0;
        __bf16* dst = &Xl[row * XSTRIDE + c0];
#pragma unroll 4
        for (int c = 0; c < 512; c += 4) {
            float4 f = *(const float4*)(src + c);
            dst[c + 0] = (__bf16)f.x;  dst[c + 1] = (__bf16)f.y;
            dst[c + 2] = (__bf16)f.z;  dst[c + 3] = (__bf16)f.w;
        }
    }
    __syncthreads();

    // ---- true-class logits for the 64 rows (threads 0..63) ----
    if (t < STILE) {
        int sg  = s0 + t;
        int tid = trueIds[b * S_ + sg];
        const float*  wr = W + (size_t)tid * D_;
        const __bf16* xr = &Xl[t * XSTRIDE];
        float acc = 0.0f;
        for (int k = 0; k < D_; k += 4) {
            float4 wf = *(const float4*)(wr + k);
            acc += (float)xr[k + 0] * wf.x + (float)xr[k + 1] * wf.y +
                   (float)xr[k + 2] * wf.z + (float)xr[k + 3] * wf.w;
        }
        float sf = (float)tid;
        float p  = __logf((sf + 2.0f) / (sf + 1.0f)) *
                   (1.0f / __logf((float)(V_ + 1)));
        tl[t] = acc + bvec[tid] - __logf(p * (float)NS_);
    }
    __syncthreads();

    // ---- per-wave streaming GEMM + online softmax ----
    const int     myTrue = trueIds[b * S_ + s0 + wave * 16 + l15];
    const __bf16* xrow   = &Xl[(wave * 16 + l15) * XSTRIDE];
    const __bf16* WgB    = Wg + (size_t)b * NPAD * D_;
    const int*    sidB   = sidP  + b * NPAD;
    const float*  beffB  = beffP + b * NPAD;
    const int     kb     = half * 8;    // A-frag K base within 32 (16-bit A layout)
    const int     xb     = half * 8;    // B-frag K base within 32 (same layout as A)

    float m = NEGBIG, lsum = 0.0f;

    for (int ntile = 0; ntile < NS_; ntile += 16) {
        // A-matrix row for this lane: pre-gathered bf16 row (tail rows are 0)
        const __bf16* wr = WgB + (size_t)(ntile + l15) * D_;

        // hoisted, vectorized epilogue metadata (32B aligned thanks to NPAD)
        const int nb  = ntile + half * 8;
        int4   si0 = *(const int4*)  (sidB  + nb);
        int4   si1 = *(const int4*)  (sidB  + nb + 4);
        float4 be0 = *(const float4*)(beffB + nb);
        float4 be1 = *(const float4*)(beffB + nb + 4);

        v8f acc = {};
#pragma unroll 4
        for (int k0 = 0; k0 < D_; k0 += 32) {
            // A fragment: bf16 chunks [kb,kb+8) and [kb+16,kb+24)
            Frag16 a;
            a.q[0] = *(const uint4*)(wr + k0 + kb);
            a.q[1] = *(const uint4*)(wr + k0 + kb + 16);

            // B fragment: bf16 chunks [xb,xb+8) and [xb+16,xb+24) of this lane's s-row from LDS
            Frag16 bb;
            const uint4* xl = (const uint4*)(xrow + k0 + xb);
            bb.q[0] = xl[0];
            bb.q[1] = xl[2];

            acc = __builtin_amdgcn_wmma_f32_16x16x32_bf16(
                      false, a.v, false, bb.v, (short)0, acc, false, false);
        }

        // branchless online logsumexp update
        // lane holds rows n = ntile + half*8 + i ; padded rows self-mask
        float vals[8];
        vals[0] = (si0.x != myTrue) ? acc[0] + be0.x : NEGBIG;
        vals[1] = (si0.y != myTrue) ? acc[1] + be0.y : NEGBIG;
        vals[2] = (si0.z != myTrue) ? acc[2] + be0.z : NEGBIG;
        vals[3] = (si0.w != myTrue) ? acc[3] + be0.w : NEGBIG;
        vals[4] = (si1.x != myTrue) ? acc[4] + be1.x : NEGBIG;
        vals[5] = (si1.y != myTrue) ? acc[5] + be1.y : NEGBIG;
        vals[6] = (si1.z != myTrue) ? acc[6] + be1.z : NEGBIG;
        vals[7] = (si1.w != myTrue) ? acc[7] + be1.w : NEGBIG;

        float vmax = vals[0];
#pragma unroll
        for (int i = 1; i < 8; ++i) vmax = fmaxf(vmax, vals[i]);

        float nm = fmaxf(fmaxf(m, vmax), -1.0e30f);   // clamp: NaN-free
        float sc = __expf(m - nm);                    // underflows to 0 safely
        float add = 0.0f;
#pragma unroll
        for (int i = 0; i < 8; ++i) add += __expf(vals[i] - nm);
        lsum = lsum * sc + add;
        m = nm;
    }

    // combine the two lane-halves (each handled half of the n rows)
    float m2 = __shfl_xor(m, 16, 32);
    float l2 = __shfl_xor(lsum, 16, 32);
    float M  = fmaxf(m, m2);
    float Lt = lsum * __expf(m - M) + l2 * __expf(m2 - M);

    // fold in the true logit, compute loss for this s
    float tt = tl[wave * 16 + l15];
    float M2 = fmaxf(M, tt);
    float Lf = Lt * __expf(M - M2) + __expf(tt - M2);
    float loss = __logf(Lf) + M2 - tt;

    // sum over the wave's 16 s-rows (both halves hold identical values)
    for (int off = 8; off >= 1; off >>= 1)
        loss += __shfl_xor(loss, off, 32);
    if (lane == 0) wsum[wave] = loss;
    __syncthreads();
    if (t == 0)
        partial[bid] = wsum[0] + wsum[1] + wsum[2] + wsum[3];
}

// ---------------------------------------------------------------------------
// Kernel 4: deterministic final reduction -> per-batch mean loss
// ---------------------------------------------------------------------------
__global__ void ssm_reduce_kernel(const float* __restrict__ partial,
                                  float* __restrict__ out) {
    int b = threadIdx.x;
    if (b < B_) {
        float s = 0.0f;
        for (int g = 0; g < 8; ++g) s += partial[b * 8 + g];
        out[b] = s * (1.0f / (float)S_);
    }
}

// ---------------------------------------------------------------------------
extern "C" void kernel_launch(void* const* d_in, const int* in_sizes, int n_in,
                              void* d_out, int out_size, void* d_ws, size_t ws_size,
                              hipStream_t stream) {
    (void)in_sizes; (void)n_in; (void)out_size; (void)ws_size;
    const float* X      = (const float*)d_in[0];   // lstm_outputs [B,S,D]
    const int*   tids   = (const int*)  d_in[1];   // next_token_ids [B,S]
    const int*   sids   = (const int*)  d_in[2];   // sampled_ids [B,NS]
    const float* W      = (const float*)d_in[3];   // softmax_W [V,D]
    const float* bvec   = (const float*)d_in[4];   // softmax_b [V]

    // workspace layout
    __bf16* Wg     = (__bf16*)d_ws;                          // B*NPAD*D bf16 (64 MB)
    float*  beffP  = (float*)(Wg + (size_t)B_ * NPAD * D_);  // B*NPAD floats
    int*    sidPtr = (int*)(beffP + (size_t)B_ * NPAD);      // B*NPAD ints
    float*  partial= (float*)(sidPtr + (size_t)B_ * NPAD);   // B*(S/STILE) floats

    ssm_meta_kernel<<<(B_ * NPAD + 255) / 256, 256, 0, stream>>>(
        sids, bvec, sidPtr, beffP);
    ssm_wconv_kernel<<<B_ * NPAD, 128, 0, stream>>>(sids, W, Wg);
    ssm_main_kernel<<<B_ * (S_ / STILE), 128, 0, stream>>>(
        X, tids, W, bvec, Wg, sidPtr, beffP, partial);
    ssm_reduce_kernel<<<1, 32, 0, stream>>>(partial, (float*)d_out);
}
